// GCN_Net_multi_linear_48524540511070
// MI455X (gfx1250) — compile-verified, matrix operand order checked by reading, not run on hardware
//
#include <hip/hip_runtime.h>
#include <hip/hip_bf16.h>

typedef __attribute__((ext_vector_type(16))) _Float16 v16h;
typedef __attribute__((ext_vector_type(8)))  float    v8f;

// ---------------- degree / normalization ----------------

__global__ void deg_init_kernel(float* deg, int n) {
    int i = blockIdx.x * blockDim.x + threadIdx.x;
    if (i < n) deg[i] = 1.0f;               // self-loop contributes 1
}

__global__ void deg_acc_kernel(const long long* __restrict__ dst, float* deg, int e) {
    int i = blockIdx.x * blockDim.x + threadIdx.x;
    if (i < e) atomicAdd(&deg[(int)dst[i]], 1.0f);
}

__global__ void dinv_kernel(float* deg_dinv, int n) {
    int i = blockIdx.x * blockDim.x + threadIdx.x;
    if (i < n) deg_dinv[i] = rsqrtf(deg_dinv[i]);   // deg >= 1 always (self-loops)
}

// Precompute per-edge int32 indices + symmetric norm once; reused by all 3 layers.
__global__ void edge_prep_kernel(const long long* __restrict__ src,
                                 const long long* __restrict__ dst,
                                 const float* __restrict__ dinv,
                                 int* __restrict__ es, int* __restrict__ ed,
                                 float* __restrict__ enrm, int e) {
    int i = blockIdx.x * blockDim.x + threadIdx.x;
    if (i < e) {
        int s = (int)src[i];
        int d = (int)dst[i];
        es[i] = s;
        ed[i] = d;
        enrm[i] = dinv[s] * dinv[d];
    }
}

// ---------------- WMMA GEMM: H[n,NOUT] = X[n,64] @ W[64,NOUT] ----------------
// Full 16-row strips only (nstrips = n/16). block = 128 threads = 4 waves;
// each wave computes one strip (all NOUT cols). No bounds checks -> branch-free stores.

template <int NOUT>
__global__ void gemm_wmma_kernel(const float* __restrict__ X,
                                 const float* __restrict__ W,
                                 float* __restrict__ H, int nstrips) {
    // Stage W transposed into LDS as f16: sW[j*64 + k] = W[k][j]
    __shared__ _Float16 sW[NOUT * 64];
    for (int idx = threadIdx.x; idx < 64 * NOUT; idx += blockDim.x) {
        int k = idx / NOUT, j = idx % NOUT;
        sW[j * 64 + k] = (_Float16)W[idx];
    }
    __syncthreads();

    const int wave  = threadIdx.x >> 5;
    const int lane  = threadIdx.x & 31;
    const int strip = blockIdx.x * 4 + wave;
    if (strip >= nstrips) return;
    const int row0 = strip * 16;

    const int m    = lane & 15;
    const int half = lane >> 4;
    const float4* xr4 = (const float4*)(X + (size_t)(row0 + m) * 64);

    // A fragments for the two K-steps (K = ks*32 .. ks*32+31)
    v16h a[2];
#pragma unroll
    for (int ks = 0; ks < 2; ++ks) {
        const int kb = ks * 32 + half * 8;            // multiple of 8
        float4 p0 = xr4[kb / 4];
        float4 p1 = xr4[kb / 4 + 1];
        float4 q0 = xr4[(kb + 16) / 4];
        float4 q1 = xr4[(kb + 16) / 4 + 1];
        a[ks][0]  = (_Float16)p0.x;  a[ks][1]  = (_Float16)p0.y;
        a[ks][2]  = (_Float16)p0.z;  a[ks][3]  = (_Float16)p0.w;
        a[ks][4]  = (_Float16)p1.x;  a[ks][5]  = (_Float16)p1.y;
        a[ks][6]  = (_Float16)p1.z;  a[ks][7]  = (_Float16)p1.w;
        a[ks][8]  = (_Float16)q0.x;  a[ks][9]  = (_Float16)q0.y;
        a[ks][10] = (_Float16)q0.z;  a[ks][11] = (_Float16)q0.w;
        a[ks][12] = (_Float16)q1.x;  a[ks][13] = (_Float16)q1.y;
        a[ks][14] = (_Float16)q1.z;  a[ks][15] = (_Float16)q1.w;
    }

    // Base for D stores: row (row0 + 8*half), col m; advance by NOUT per r.
    float* hbase = H + (size_t)(row0 + 8 * half) * NOUT + m;

#pragma unroll
    for (int nt = 0; nt < NOUT / 16; ++nt) {
        v8f c = {};
        const _Float16* wc = &sW[(nt * 16 + m) * 64];
#pragma unroll
        for (int ks = 0; ks < 2; ++ks) {
            const int kb = ks * 32 + half * 8;
            v16h b;
#pragma unroll
            for (int i = 0; i < 8; ++i) {
                b[i]     = wc[kb + i];
                b[i + 8] = wc[kb + 16 + i];
            }
            c = __builtin_amdgcn_wmma_f32_16x16x32_f16(
                    false, a[ks], false, b, (short)0, c, false, false);
        }
        // D layout: VGPR r -> row = row0 + r + 8*half, col = nt*16 + m
#pragma unroll
        for (int r = 0; r < 8; ++r)
            hbase[(size_t)r * NOUT + nt * 16] = c[r];
    }
}

// Scalar tail for rows [nfull*16, n); launched only when n % 16 != 0.
template <int NOUT>
__global__ void gemm_tail_kernel(const float* __restrict__ X,
                                 const float* __restrict__ W,
                                 float* __restrict__ H, int rowstart, int n) {
    int t = blockIdx.x * blockDim.x + threadIdx.x;
    int row = rowstart + t / NOUT;
    int col = t % NOUT;
    if (row >= n) return;
    float acc = 0.f;
#pragma unroll
    for (int k = 0; k < 64; ++k)
        acc += X[(size_t)row * 64 + k] * W[(size_t)k * NOUT + col];
    H[(size_t)row * NOUT + col] = acc;
}

// ---------------- edge scatter: agg[dst] += h[src] * norm[e] ----------------
// D/4 threads per edge, float4 gather + 4 float atomics each (contiguous, L2-resident).

template <int D>
__global__ void scatter_kernel(const int* __restrict__ es,
                               const int* __restrict__ ed,
                               const float* __restrict__ enrm,
                               const float* __restrict__ H,
                               float* agg, int E) {
    const int CPE = D / 4;
    long long gid = (long long)blockIdx.x * blockDim.x + threadIdx.x;
    int e = (int)(gid / CPE);
    if (e >= E) return;
    int c = (int)(gid % CPE) * 4;
    int s = es[e];
    int d = ed[e];
    float nrm = enrm[e];
    float4 hv = *(const float4*)(H + (size_t)s * D + c);
    float* outp = agg + (size_t)d * D + c;
    atomicAdd(outp + 0, hv.x * nrm);
    atomicAdd(outp + 1, hv.y * nrm);
    atomicAdd(outp + 2, hv.z * nrm);
    atomicAdd(outp + 3, hv.w * nrm);
}

// ---------------- finalize: out = [relu](agg + h*dinv^2 + b) (in place OK) ----------------

template <int D, bool RELU>
__global__ void finalize_kernel(const float* __restrict__ agg,
                                const float* __restrict__ H,
                                const float* __restrict__ dinv,
                                const float* __restrict__ bias,
                                float* out, int n) {
    long long gid = (long long)blockIdx.x * blockDim.x + threadIdx.x;
    int i = (int)(gid / D);
    if (i >= n) return;
    int j = (int)(gid % D);
    float di = dinv[i];
    float v = agg[(size_t)i * D + j] + H[(size_t)i * D + j] * di * di + bias[j];
    if (RELU) v = fmaxf(v, 0.0f);
    out[(size_t)i * D + j] = v;
}

// ---------------- launch ----------------

extern "C" void kernel_launch(void* const* d_in, const int* in_sizes, int n_in,
                              void* d_out, int out_size, void* d_ws, size_t ws_size,
                              hipStream_t stream) {
    const float*     x  = (const float*)d_in[0];
    const long long* ei = (const long long*)d_in[1];
    const float*     W1 = (const float*)d_in[2];
    const float*     b1 = (const float*)d_in[3];
    const float*     W2 = (const float*)d_in[4];
    const float*     b2 = (const float*)d_in[5];
    const float*     W3 = (const float*)d_in[6];
    const float*     b3 = (const float*)d_in[7];
    float* out = (float*)d_out;

    const int n = in_sizes[0] / 64;     // 50000
    const int e = in_sizes[1] / 2;      // 800000
    const long long* src = ei;
    const long long* dst = ei + e;

    // workspace layout
    char* ws = (char*)d_ws;
    size_t off = 0;
    auto alloc = [&](size_t bytes) {
        void* p = ws + off;
        off += (bytes + 255) & ~(size_t)255;
        return p;
    };
    float* dinv = (float*)alloc((size_t)n * 4);
    float* h    = (float*)alloc((size_t)n * 64 * 4);
    float* xa   = (float*)alloc((size_t)n * 64 * 4);
    int*   es   = (int*)  alloc((size_t)e * 4);
    int*   ed   = (int*)  alloc((size_t)e * 4);
    float* enrm = (float*)alloc((size_t)e * 4);

    const int T = 256;
    auto nb = [](long long w, int t) { return (int)((w + t - 1) / t); };

    // normalization + per-edge norm table (computed once, reused by all 3 layers)
    deg_init_kernel<<<nb(n, T), T, 0, stream>>>(dinv, n);
    deg_acc_kernel<<<nb(e, T), T, 0, stream>>>(dst, dinv, e);
    dinv_kernel<<<nb(n, T), T, 0, stream>>>(dinv, n);
    edge_prep_kernel<<<nb(e, T), T, 0, stream>>>(src, dst, dinv, es, ed, enrm, e);

    const int nstrips = n / 16;                 // full strips (n=50000 -> 3125, exact)
    const int tail    = n - nstrips * 16;       // 0 for this workload
    const int gblocks = (nstrips + 3) / 4;

    auto gemm = [&](auto noutTag, const float* X, const float* W, float* H) {
        constexpr int NOUT = decltype(noutTag)::value;
        gemm_wmma_kernel<NOUT><<<gblocks, 128, 0, stream>>>(X, W, H, nstrips);
        if (tail > 0)   // host-side, deterministic (n fixed per harness)
            gemm_tail_kernel<NOUT><<<nb((long long)tail * NOUT, T), T, 0, stream>>>(
                X, W, H, nstrips * 16, n);
    };

    // ---- layer 1: x -> xa (relu) ----
    gemm(std::integral_constant<int, 64>{}, x, W1, h);
    hipMemsetAsync(xa, 0, (size_t)n * 64 * 4, stream);
    scatter_kernel<64><<<nb((long long)e * 16, T), T, 0, stream>>>(es, ed, enrm, h, xa, e);
    finalize_kernel<64, true><<<nb((long long)n * 64, T), T, 0, stream>>>(xa, h, dinv, b1, xa, n);

    // ---- layer 2: xa -> xa (relu) ----
    gemm(std::integral_constant<int, 64>{}, xa, W2, h);
    hipMemsetAsync(xa, 0, (size_t)n * 64 * 4, stream);   // gemm done reading xa (stream-ordered)
    scatter_kernel<64><<<nb((long long)e * 16, T), T, 0, stream>>>(es, ed, enrm, h, xa, e);
    finalize_kernel<64, true><<<nb((long long)n * 64, T), T, 0, stream>>>(xa, h, dinv, b2, xa, n);

    // ---- layer 3: xa -> d_out (no relu), D = 16 ----
    gemm(std::integral_constant<int, 16>{}, xa, W3, h);
    hipMemsetAsync(out, 0, (size_t)n * 16 * 4, stream);
    scatter_kernel<16><<<nb((long long)e * 4, T), T, 0, stream>>>(es, ed, enrm, h, out, e);
    finalize_kernel<16, false><<<nb((long long)n * 16, T), T, 0, stream>>>(out, h, dinv, b3, out, n);
}
